// SurnameClassifier_49701361549664
// MI455X (gfx1250) — compile-verified, matrix operand order checked by reading, not run on hardware
//
#include <hip/hip_runtime.h>
#include <hip/hip_bf16.h>
#include <stdint.h>

// ---------------------------------------------------------------------------
// SurnameClassifier on MI455X (gfx1250).
//  - prologue: convert emb/weights f32 -> bf16 in workspace
//  - x_proj GEMM (embedding gather fused), 64 sequential RNN step GEMMs
//    (tanh + last-hidden gather fused), fc1 (relu), fc2.
//  - all GEMMs: v_wmma_f32_16x16x32_bf16, double-buffered LDS tiles filled
//    with global_load_async_to_lds_b128 (ASYNCcnt), fragments via ds_load_b128
//    in an explicit 1-deep software pipeline so DS latency hides under WMMA.
// ---------------------------------------------------------------------------

typedef __attribute__((ext_vector_type(8)))  __bf16 v8bf;
typedef __attribute__((ext_vector_type(16))) __bf16 v16bf;
typedef __attribute__((ext_vector_type(8)))  float  v8f;

#define LDT 40  // LDS row stride in bf16 elems (32 data + 8 pad; rows 16B-aligned)

__device__ inline void async_copy16(uint32_t lds_off, const void* gptr) {
  // GLOBAL_LOAD_ASYNC_TO_LDS_B128: 16 bytes per lane, tracked by ASYNCcnt.
  asm volatile("global_load_async_to_lds_b128 %0, %1, off"
               :: "v"(lds_off), "v"(gptr)
               : "memory");
}

__device__ inline void wait_async0() {
  asm volatile("s_wait_asynccnt 0x0" ::: "memory");
}

// Fragment load per CDNA5 16-bit 16x32 layout: lanes 0-15 hold K in
// {0..7, 16..23}, lanes 16-31 hold K in {8..15, 24..31}.
__device__ inline v16bf frag_load(const __bf16* rowbase, int half) {
  v8bf lo = *(const v8bf*)(rowbase + 8 * half);
  v8bf hi = *(const v8bf*)(rowbase + 16 + 8 * half);
  return __builtin_shufflevector(lo, hi, 0, 1, 2, 3, 4, 5, 6, 7,
                                 8, 9, 10, 11, 12, 13, 14, 15);
}

// MODE 0: x_proj : A = gather(emb_bf, x_in) [M=S*B, K=E], +bias -> f32 out
// MODE 1: step   : A = h_prev bf16, +bias +xproj_t(f32), tanh -> h_new bf16,
//                  ylast[b] = h_new[b] where lens[b]-1 == t
// MODE 2: fc1    : +bias, relu -> bf16 out
// MODE 3: fc2    : +bias -> f32 out, N bounds-checked
// GEMM: C[m,n] = sum_k A[m,k] * W[n,k]  (A @ W^T), all operands bf16.
template <int MODE, int BT>
__global__ __launch_bounds__(32 * BT) void gemm_wmma_kernel(
    const __bf16* __restrict__ Aglob,
    const int* __restrict__ x_in,       // MODE==0, [B,S]
    const __bf16* __restrict__ emb,     // MODE==0, [V,E] bf16
    const __bf16* __restrict__ Wglob,   // [N,K] bf16
    const float* __restrict__ bias,     // [N]
    const float* __restrict__ xproj_t,  // MODE==1, [M,N] f32 slice at step t
    const int* __restrict__ lens,       // MODE==1, [B]
    __bf16* __restrict__ ylast_bf,      // MODE==1, [B,H]
    float* __restrict__ Cout_f,         // MODE 0/3
    __bf16* __restrict__ Cout_bf,       // MODE 1/2
    int M, int N, int K, int t, int Bdim, int Sdim) {
  constexpr int TM = BT * 16;
  const int tid  = threadIdx.x;
  const int wave = tid >> 5;
  const int lane = tid & 31;
  const int half = lane >> 4;
  const int l16  = lane & 15;
  const int row0 = blockIdx.x * TM;
  const int col0 = blockIdx.y * TM;

  __shared__ __align__(16) __bf16 As[2][TM * LDT];
  __shared__ __align__(16) __bf16 Bs[2][TM * LDT];
  __shared__ int tok[TM];

  const uint32_t As_base = (uint32_t)(uintptr_t)(&As[0][0]);
  const uint32_t Bs_base = (uint32_t)(uintptr_t)(&Bs[0][0]);

  if (MODE == 0) {
    if (tid < TM) {
      int m = row0 + tid;
      int s = m / Bdim;
      int b = m - s * Bdim;
      tok[tid] = x_in[b * Sdim + s];
    }
    __syncthreads();  // tok visible before async address generation
  }

  // Issue one tile (TM x 32 bf16 = TM*64B) as 16B-per-lane async copies.
  // chunks = TM*4, threads = 2*TM -> 2 chunks per thread.
  auto issueA = [&](int k0, int buf) {
#pragma unroll
    for (int i = 0; i < 2; ++i) {
      int e  = tid + i * (32 * BT);
      int r  = e >> 2;
      int c8 = (e & 3) * 8;
      const __bf16* src;
      if (MODE == 0)
        src = emb + (size_t)tok[r] * K + k0 + c8;
      else
        src = Aglob + (size_t)(row0 + r) * K + k0 + c8;
      async_copy16(As_base + 2u * (uint32_t)(buf * TM * LDT + r * LDT + c8), src);
    }
  };
  auto issueB = [&](int k0, int buf) {
#pragma unroll
    for (int i = 0; i < 2; ++i) {
      int e  = tid + i * (32 * BT);
      int r  = e >> 2;
      int c8 = (e & 3) * 8;
      int n  = col0 + r;
      if (MODE == 3 && n > N - 1) n = N - 1;  // clamp: dead cols never stored
      const __bf16* src = Wglob + (size_t)n * K + k0 + c8;
      async_copy16(Bs_base + 2u * (uint32_t)(buf * TM * LDT + r * LDT + c8), src);
    }
  };

  v8f acc[BT];
#pragma unroll
  for (int j = 0; j < BT; ++j) acc[j] = v8f{};

  const int nk = K / 32;
  issueA(0, 0);
  issueB(0, 0);

  for (int ki = 0; ki < nk; ++ki) {
    const int buf = ki & 1;
    wait_async0();     // this wave's async writes to LDS are complete
    __syncthreads();   // all waves' writes visible / prior reads retired

    if (ki + 1 < nk) {  // prefetch next chunk into the other buffer
      issueA((ki + 1) * 32, buf ^ 1);
      issueB((ki + 1) * 32, buf ^ 1);
    }

    // A fragment + 1-deep pipelined B fragments: the DS pair for fragment
    // j+1 is issued before WMMA j, so its latency hides under the matrix op.
    v16bf afrag = frag_load(&As[buf][(wave * 16 + l16) * LDT], half);
    v16bf bcur  = frag_load(&Bs[buf][(l16)*LDT], half);
#pragma unroll
    for (int j = 0; j < BT; ++j) {
      v16bf bnext;
      if (j + 1 < BT)
        bnext = frag_load(&Bs[buf][((j + 1) * 16 + l16) * LDT], half);
      acc[j] = __builtin_amdgcn_wmma_f32_16x16x32_bf16(
          false, afrag, false, bcur, (short)0, acc[j], false, false);
      if (j + 1 < BT) bcur = bnext;
    }
  }

  // Epilogue. C frag layout: VGPR r -> m_local = 8*half + r, n_local = l16.
  const int mbase = row0 + wave * 16 + half * 8;
#pragma unroll
  for (int j = 0; j < BT; ++j) {
    int n = col0 + j * 16 + l16;
    if (MODE == 3 && n >= N) continue;
    float bn = bias[n];
#pragma unroll
    for (int r = 0; r < 8; ++r) {
      int mm  = mbase + r;
      float v = acc[j][r] + bn;
      if (MODE == 0) {
        Cout_f[(size_t)mm * N + n] = v;
      } else if (MODE == 1) {
        v = tanhf(v + xproj_t[(size_t)mm * N + n]);
        Cout_bf[(size_t)mm * N + n] = (__bf16)v;
        if (lens[mm] - 1 == t) ylast_bf[(size_t)mm * N + n] = (__bf16)v;
      } else if (MODE == 2) {
        Cout_bf[(size_t)mm * N + n] = (__bf16)fmaxf(v, 0.f);
      } else {
        Cout_f[(size_t)mm * N + n] = v;
      }
    }
  }
}

__global__ void cvt_f32_bf16_kernel(const float* __restrict__ src,
                                    __bf16* __restrict__ dst, int n4) {
  int i = blockIdx.x * blockDim.x + threadIdx.x;
  if (i < n4) {
    float4 v = ((const float4*)src)[i];
    __bf16* d = dst + 4 * (size_t)i;
    d[0] = (__bf16)v.x; d[1] = (__bf16)v.y;
    d[2] = (__bf16)v.z; d[3] = (__bf16)v.w;
  }
}

__global__ void zero_u32_kernel(uint32_t* __restrict__ p, int n) {
  int i = blockIdx.x * blockDim.x + threadIdx.x;
  if (i < n) p[i] = 0u;
}

extern "C" void kernel_launch(void* const* d_in, const int* in_sizes, int n_in,
                              void* d_out, int out_size, void* d_ws,
                              size_t ws_size, hipStream_t stream) {
  (void)in_sizes; (void)n_in; (void)out_size; (void)ws_size;
  constexpr int B = 512, S = 64, E = 512, H = 1024, V = 1024, C = 1000;

  const int*   x_in  = (const int*)d_in[0];
  const int*   x_len = (const int*)d_in[1];
  const float* emb_w = (const float*)d_in[2];
  const float* w_ih  = (const float*)d_in[3];
  const float* b_ih  = (const float*)d_in[4];
  const float* w_hh  = (const float*)d_in[5];
  const float* b_hh  = (const float*)d_in[6];
  const float* fc1_w = (const float*)d_in[7];
  const float* fc1_b = (const float*)d_in[8];
  const float* fc2_w = (const float*)d_in[9];
  const float* fc2_b = (const float*)d_in[10];

  // Workspace layout
  float*  xproj  = (float*)d_ws;                      // [S,B,H] f32 (128 MB)
  __bf16* h0     = (__bf16*)(xproj + (size_t)S * B * H);
  __bf16* h1     = h0 + (size_t)B * H;
  __bf16* ylast  = h1 + (size_t)B * H;
  __bf16* act    = ylast + (size_t)B * H;
  __bf16* emb_bf = act + (size_t)B * H;               // [V,E]
  __bf16* wih_bf = emb_bf + (size_t)V * E;            // [H,E]
  __bf16* whh_bf = wih_bf + (size_t)H * E;            // [H,H]
  __bf16* f1w_bf = whh_bf + (size_t)H * H;            // [H,H]
  __bf16* f2w_bf = f1w_bf + (size_t)H * H;            // [C,H]

  // Prologue: f32 -> bf16 weight conversions; zero h0.
  auto cvt = [&](const float* s, __bf16* d, size_t n) {
    int n4 = (int)(n / 4);
    cvt_f32_bf16_kernel<<<(n4 + 255) / 256, 256, 0, stream>>>(s, d, n4);
  };
  cvt(emb_w, emb_bf, (size_t)V * E);
  cvt(w_ih, wih_bf, (size_t)H * E);
  cvt(w_hh, whh_bf, (size_t)H * H);
  cvt(fc1_w, f1w_bf, (size_t)H * H);
  cvt(fc2_w, f2w_bf, (size_t)C * H);
  zero_u32_kernel<<<(B * H / 2 + 255) / 256, 256, 0, stream>>>(
      (uint32_t*)h0, B * H / 2);

  // x_proj[s,b,:] = emb[x_in[b,s]] @ w_ih^T + b_ih  (M = S*B = 32768)
  gemm_wmma_kernel<0, 8><<<dim3((S * B) / 128, H / 128), 256, 0, stream>>>(
      nullptr, x_in, emb_bf, wih_bf, b_ih, nullptr, nullptr, nullptr, xproj,
      nullptr, S * B, H, E, 0, B, S);

  // RNN scan: h = tanh(x_proj[t] + h @ w_hh^T + b_hh); fused y_last gather.
  __bf16* hin = h0;
  __bf16* hout = h1;
  for (int t = 0; t < S; ++t) {
    gemm_wmma_kernel<1, 4><<<dim3(B / 64, H / 64), 128, 0, stream>>>(
        hin, nullptr, nullptr, whh_bf, b_hh, xproj + (size_t)t * B * H, x_len,
        ylast, nullptr, hout, B, H, H, t, B, S);
    __bf16* tmp = hin; hin = hout; hout = tmp;
  }

  // fc1: act = relu(y_last @ fc1_w^T + fc1_b)
  gemm_wmma_kernel<2, 4><<<dim3(B / 64, H / 64), 128, 0, stream>>>(
      ylast, nullptr, nullptr, f1w_bf, fc1_b, nullptr, nullptr, nullptr,
      nullptr, act, B, H, H, 0, B, S);

  // fc2: out = act @ fc2_w^T + fc2_b  (N = 1000)
  gemm_wmma_kernel<3, 4><<<dim3(B / 64, (C + 63) / 64), 128, 0, stream>>>(
      act, nullptr, nullptr, f2w_bf, fc2_b, nullptr, nullptr, nullptr,
      (float*)d_out, nullptr, B, C, H, 0, B, S);
}
